// InteractionPruningLayer_72378788872995
// MI455X (gfx1250) — matrix operand order repaired and checked by reading, not compiled, not run on hardware
//
#include <hip/hip_runtime.h>

#define B_ 1024
#define F_ 256
#define D_ 64

typedef __attribute__((ext_vector_type(16))) _Float16 v16h;
typedef __attribute__((ext_vector_type(8)))  _Float16 v8h;
typedef __attribute__((ext_vector_type(8)))  float    v8f;

// ---------------- Kernel A1: qkv5[n][f][e] for n in {trans0,trans1,qkv0,qkv1,qkv2} --
__global__ __launch_bounds__(256) void kA1(const float* __restrict__ ind,
                                           const float* __restrict__ Wqk,
                                           const float* __restrict__ Wqkv,
                                           float* __restrict__ qkv5) {
  int idx = blockIdx.x * 256 + threadIdx.x;   // 5*F*D = 81920 threads exactly
  int n = idx >> 14;                          // / (F_*D_)
  int r = idx & 16383;
  int f = r >> 6, e = r & 63;
  const float* W = (n < 2) ? (Wqk + n * D_ * D_) : (Wqkv + (n - 2) * D_ * D_);
  const float* irow = ind + f * D_;
  float acc = 0.f;
#pragma unroll 8
  for (int d = 0; d < D_; ++d) acc += irow[d] * W[d * D_ + e];
  qkv5[idx] = acc;
}

// ---------------- Kernel A2: M[i][j] = gate(i,j) ? cross(i,j) : 0, stored f16 -------
__global__ __launch_bounds__(256) void kA2(const float* __restrict__ qkv5,
                                           _Float16* __restrict__ Mh) {
  int idx = blockIdx.x * 256 + threadIdx.x;   // F*F = 65536 threads exactly
  int i = idx >> 8, j = idx & 255;
  const float4* t0 = (const float4*)(qkv5 + 0 * F_ * D_ + i * D_);
  const float4* t1 = (const float4*)(qkv5 + 1 * F_ * D_ + j * D_);
  const float4* q1 = (const float4*)(qkv5 + 3 * F_ * D_ + i * D_);
  const float4* q0 = (const float4*)(qkv5 + 2 * F_ * D_ + j * D_);
  float dg = 0.f, dc = 0.f;
#pragma unroll
  for (int d4 = 0; d4 < D_ / 4; ++d4) {
    float4 a = t0[d4], b = t1[d4], c = q1[d4], e = q0[d4];
    dg += a.x * b.x + a.y * b.y + a.z * b.z + a.w * b.w;
    dc += c.x * e.x + c.y * e.y + c.z * e.z + c.w * e.w;
  }
  Mh[idx] = (dg > 0.f) ? (_Float16)dc : (_Float16)0.f;
}

// ---------------- Kernel B: g[b,f] = s0*s2 (f32) ; s1h[b,f] (f16) -------------------
__global__ __launch_bounds__(256) void kB(const float* __restrict__ feat,
                                          const float* __restrict__ qkv5,
                                          float* __restrict__ g,
                                          _Float16* __restrict__ s1h) {
  int idx = blockIdx.x * 256 + threadIdx.x;   // B*F = 262144 threads exactly
  int f = idx & 255;
  const float4* fr = (const float4*)(feat + (size_t)idx * D_);
  const float4* q0 = (const float4*)(qkv5 + 2 * F_ * D_ + f * D_);
  const float4* q1 = (const float4*)(qkv5 + 3 * F_ * D_ + f * D_);
  const float4* q2 = (const float4*)(qkv5 + 4 * F_ * D_ + f * D_);
  float a0 = 0.f, a1 = 0.f, a2 = 0.f;
#pragma unroll
  for (int d4 = 0; d4 < D_ / 4; ++d4) {
    float4 x = fr[d4];
    float4 y0 = q0[d4], y1 = q1[d4], y2 = q2[d4];
    a0 += x.x * y0.x + x.y * y0.y + x.z * y0.z + x.w * y0.w;
    a1 += x.x * y1.x + x.y * y1.y + x.z * y1.z + x.w * y1.w;
    a2 += x.x * y2.x + x.y * y2.y + x.z * y2.z + x.w * y2.w;
  }
  g[idx]   = a0 * a2;
  s1h[idx] = (_Float16)a1;
}

// ---------------- Kernel Q2T: q2t[d][f] = f16(qkv2[f][d])  (batch-invariant, 32KB) --
__global__ __launch_bounds__(256) void kQ2T(const float* __restrict__ qkv5,
                                            _Float16* __restrict__ q2t) {
  int idx = blockIdx.x * 256 + threadIdx.x;   // D*F = 16384 threads exactly
  int f = idx & 255;
  int d = idx >> 8;
  q2t[idx] = (_Float16)qkv5[4 * F_ * D_ + f * D_ + d];
}

// ---------------- Kernel C: out[b] = diag(g[b]) * (M @ (s1[b] (*) q2t)) via WMMA ----
// A fragment (16x32 f16): lane&15 = row; K = K0+16*half+s (contiguous 16)
// -> two contiguous 16B loads.
__device__ __forceinline__ v16h load_a16(const _Float16* row, int k0) {
  v8h lo = *(const v8h*)(row + k0);
  v8h hi = *(const v8h*)(row + k0 + 8);
  v16h r;
#pragma unroll
  for (int i = 0; i < 8; ++i) { r[i] = lo[i]; r[8 + i] = hi[i]; }
  return r;
}

__global__ __launch_bounds__(256) void kC(const _Float16* __restrict__ Mh,
                                          const _Float16* __restrict__ q2t,
                                          const _Float16* __restrict__ s1h,
                                          const float* __restrict__ g,
                                          float* __restrict__ out) {
  const int b    = blockIdx.x;
  const int tid  = threadIdx.x;
  const int w    = tid >> 5;            // wave id 0..7
  const int lane = tid & 31;
  const int kh   = lane >> 4;           // half-wave
  const int l15  = lane & 15;

  // ---- Stage q2t (32KB, [d][f] f16) into LDS with CDNA5 async-to-LDS copies ----
  __shared__ __align__(16) _Float16 q2s[D_ * F_];
  {
    const char* gsrc  = (const char*)q2t;
    uint32_t    lbase = (uint32_t)(uintptr_t)q2s;
#pragma unroll
    for (int i = 0; i < 8; ++i) {                    // 2048 x 16B chunks / 256 threads
      uint32_t off = (uint32_t)(tid + 256 * i) * 16u;
      uint64_t ga  = (uint64_t)(uintptr_t)(gsrc + off);
      uint32_t la  = lbase + off;
      asm volatile("global_load_async_to_lds_b128 %0, %1, off"
                   :: "v"(la), "v"(ga) : "memory");
    }
    asm volatile("s_wait_asynccnt 0x0" ::: "memory");
    __syncthreads();
  }

  const v8f zero = {0.f, 0.f, 0.f, 0.f, 0.f, 0.f, 0.f, 0.f};
  v8f acc[2][4];
#pragma unroll
  for (int i = 0; i < 2; ++i)
#pragma unroll
    for (int c = 0; c < 4; ++c) acc[i][c] = zero;

  const int rbase = 32 * w;                               // this wave's 32 output rows
  const _Float16* Ar0  = Mh + (rbase + l15) * F_;         // row tile 0
  const _Float16* Ar1  = Mh + (rbase + 16 + l15) * F_;    // row tile 1
  const _Float16* s1b  = s1h + b * F_;

  for (int kt = 0; kt < 8; ++kt) {                        // K = 256, 32 per step
    const int K0 = kt * 32;
    const int ks = K0 + 16 * kh;                          // this lane's 16 K values
    v16h a0 = load_a16(Ar0, ks);
    v16h a1 = load_a16(Ar1, ks);

    // s1 segment (same for all 4 col tiles): 16 contiguous f16
    v16h s1v;
    {
      v8h lo = *(const v8h*)(s1b + ks);
      v8h hi = *(const v8h*)(s1b + ks + 8);
#pragma unroll
      for (int i = 0; i < 8; ++i) { s1v[i] = lo[i]; s1v[8 + i] = hi[i]; }
    }

    // B fragments built on the fly: bf[c][k] = s1[k] * q2t[n=16c+l15][k]  (LDS reads)
    v16h bf[4];
#pragma unroll
    for (int c = 0; c < 4; ++c) {
      const _Float16* qrow = q2s + (16 * c + l15) * F_ + ks;
      v16h qv;
      v8h lo = *(const v8h*)(qrow);
      v8h hi = *(const v8h*)(qrow + 8);
#pragma unroll
      for (int i = 0; i < 8; ++i) { qv[i] = lo[i]; qv[8 + i] = hi[i]; }
      bf[c] = s1v * qv;                                   // v_pk_mul_f16 x8
    }

#pragma unroll
    for (int c = 0; c < 4; ++c) {
      acc[0][c] = __builtin_amdgcn_wmma_f32_16x16x32_f16(
          false, a0, false, bf[c], (short)0, acc[0][c], false, false);
      acc[1][c] = __builtin_amdgcn_wmma_f32_16x16x32_f16(
          false, a1, false, bf[c], (short)0, acc[1][c], false, false);
    }
  }

  // Epilogue: scale row i by g[b,i] and store. C layout: row = v + 8*half, col = l15.
  float gv = g[b * F_ + rbase + lane];                    // rows rbase..rbase+31
#pragma unroll
  for (int i = 0; i < 2; ++i) {
#pragma unroll
    for (int v = 0; v < 8; ++v) {
      const int rloc = 16 * i + v + 8 * kh;               // 0..31 within strip
      const float gs = __shfl(gv, rloc, 32);
      const int row = rbase + rloc;
#pragma unroll
      for (int c = 0; c < 4; ++c)
        out[((size_t)b * F_ + row) * D_ + 16 * c + l15] = acc[i][c][v] * gs;
    }
  }
}

// ---------------- launcher ----------------------------------------------------------
extern "C" void kernel_launch(void* const* d_in, const int* in_sizes, int n_in,
                              void* d_out, int out_size, void* d_ws, size_t ws_size,
                              hipStream_t stream) {
  (void)in_sizes; (void)n_in; (void)out_size; (void)ws_size;
  const float* feature   = (const float*)d_in[0];
  const float* indicator = (const float*)d_in[1];
  const float* Wqk       = (const float*)d_in[2];
  const float* Wqkv      = (const float*)d_in[3];
  float* out = (float*)d_out;

  char* ws = (char*)d_ws;
  float*    qkv5 = (float*)(ws);                 //  327680 B : [5][F][D] f32
  _Float16* Mh   = (_Float16*)(ws + 327680);     //  131072 B : [F][F]    f16
  float*    g    = (float*)(ws + 458752);        // 1048576 B : [B][F]    f32
  _Float16* s1h  = (_Float16*)(ws + 1507328);    //  524288 B : [B][F]    f16
  _Float16* q2t  = (_Float16*)(ws + 2031616);    //   32768 B : [D][F]    f16

  kA1 <<<(5 * F_ * D_) / 256, 256, 0, stream>>>(indicator, Wqk, Wqkv, qkv5);
  kA2 <<<(F_ * F_) / 256,     256, 0, stream>>>(qkv5, Mh);
  kB  <<<(B_ * F_) / 256,     256, 0, stream>>>(feature, qkv5, g, s1h);
  kQ2T<<<(D_ * F_) / 256,     256, 0, stream>>>(qkv5, q2t);
  kC  <<<B_,                  256, 0, stream>>>(Mh, q2t, s1h, g, out);
}
